// AFNO_90975997264325
// MI455X (gfx1250) — matrix operand order repaired and checked by reading, not compiled, not run on hardware
//
#include <hip/hip_runtime.h>
#include <math.h>

// ---- static problem config (must match reference) ----
#define BB    4
#define HH    256
#define WW    256
#define TIN   10
#define FF    4
#define PP    8
#define CINC  12
#define EDIM  768
#define NBLK  8
#define BSZ   96
#define HPD   32
#define WPD   32
#define WFD   17          // 32/2 + 1
#define MIDD  3072
#define DD    4

#define NTOK  (BB*HPD*WPD)        // 4096 tokens
#define NSPEC (BB*HPD*WFD)        // 2176 spectral rows

typedef __bf16 bf16_t;
typedef __attribute__((ext_vector_type(16))) __bf16 bf16x16;
typedef __attribute__((ext_vector_type(8)))  __bf16 bf16x8;
typedef __attribute__((ext_vector_type(4)))  __bf16 bf16x4;
typedef __attribute__((ext_vector_type(8)))  float  f32x8;

__device__ __forceinline__ float gelu_erf(float v) {
    return 0.5f * v * (1.0f + erff(v * 0.70710678118654752f));
}
__device__ __forceinline__ float softshrink(float v) {
    float a = fabsf(v) - 0.01f;
    a = a > 0.0f ? a : 0.0f;
    return v > 0.0f ? a : -a;
}
__device__ __forceinline__ bf16x16 cat8(bf16x8 lo, bf16x8 hi) {
    return __builtin_shufflevector(lo, hi, 0,1,2,3,4,5,6,7,8,9,10,11,12,13,14,15);
}

// =====================================================================
// bf16 WMMA GEMM, f32 accumulate, templated wave tile (waves laid out 4x2).
//   C = A1@B1 (+ alpha2 * A2@B2) (+ bias[n]) (+ pos[m%1024,n]) -> act -> Cf/Cb
// Big config <32,32,false>: block 128x64, 4 WMMA / wave / K-step.
// Small dual config <16,16,true>: block 64x32, complex spectral MLP fused:
//   both products accumulated in separate WMMA accumulators, combined in
//   epilogue (bf16 WMMA has no input negate, so alpha2 handles the -1).
// Requires M%BM==0, N%BN==0, K%32==0 (true at every call site).
// A fragment per ISA 7.12.2: per lane two contiguous 8-elem runs -> 2x ds b128.
// B staged transposed in LDS -> B fragment is also 2x ds b128 per lane.
// grid.z batches the 8 spectral diagonal blocks in one launch.
// =====================================================================
template<int WM, int WN, bool DUAL>
__global__ __launch_bounds__(256)
void wmma_gemm_kernel(const bf16_t* __restrict__ A1, const bf16_t* __restrict__ A2,
                      long long sAb, int lda,
                      const bf16_t* __restrict__ B1, const bf16_t* __restrict__ B2,
                      long long sBb, int ldb, float alpha2,
                      float* __restrict__ Cf, bf16_t* __restrict__ Cb,
                      long long sCb, int ldc,
                      const float* __restrict__ bias, long long sBias,
                      const float* __restrict__ pos,
                      int K, int act)
{
    constexpr int BM = 4 * WM, BN = 2 * WN;
    constexpr int MI = WM / 16, NI = WN / 16;
    constexpr int LSTR = 40;                    // LDS row stride (80B, conflict-padded)

    __shared__ bf16_t As [BM][LSTR];            // [m][k], row-major
    __shared__ bf16_t Bst[BN][LSTR];            // [n][k], transposed at fill

    const int tid  = threadIdx.x;
    const int lane = tid & 31;
    const int wid  = tid >> 5;
    const int wm0  = (wid & 3) * WM;
    const int wn0  = (wid >> 2) * WN;
    const int rowBase = blockIdx.y * BM;
    const int colBase = blockIdx.x * BN;
    const long long z = blockIdx.z;

    const f32x8 z8 = {0.f,0.f,0.f,0.f,0.f,0.f,0.f,0.f};
    f32x8 acc [MI][NI];
    f32x8 acc2[MI][NI];
    #pragma unroll
    for (int i = 0; i < MI; ++i)
        #pragma unroll
        for (int j = 0; j < NI; ++j) { acc[i][j] = z8; if constexpr (DUAL) acc2[i][j] = z8; }

    #pragma unroll
    for (int phase = 0; phase < (DUAL ? 2 : 1); ++phase) {
        const bf16_t* Ap = ((DUAL && phase) ? A2 : A1) + z * sAb;
        const bf16_t* Bp = ((DUAL && phase) ? B2 : B1) + z * sBb;

        for (int k0 = 0; k0 < K; k0 += 32) {
            // ---- cooperative fill: A tile (BM x 32), b128 per chunk ----
            #pragma unroll
            for (int i = 0; i < BM / 64; ++i) {
                int cid = tid + i * 256;
                int ar = cid >> 2, acg = (cid & 3) << 3;
                const bf16_t* src = Ap + (long long)(rowBase + ar) * lda + (k0 + acg);
                if (k0 + 32 < K) __builtin_prefetch(src + 32, 0, 3);
                *(bf16x8*)&As[ar][acg] = *(const bf16x8*)src;
            }
            // ---- cooperative fill: B tile (32 x BN), stored transposed ----
            if constexpr (BN == 64) {
                int br = tid >> 3, bcg = (tid & 7) << 3;
                const bf16_t* src = Bp + (long long)(k0 + br) * ldb + (colBase + bcg);
                if (k0 + 32 < K) __builtin_prefetch(src + (long long)32 * ldb, 0, 3);
                bf16x8 v = *(const bf16x8*)src;
                #pragma unroll
                for (int j = 0; j < 8; ++j) Bst[bcg + j][br] = v[j];
            } else {
                int br = tid >> 3, bcg = (tid & 7) << 2;
                const bf16_t* src = Bp + (long long)(k0 + br) * ldb + (colBase + bcg);
                if (k0 + 32 < K) __builtin_prefetch(src + (long long)32 * ldb, 0, 3);
                bf16x4 v = *(const bf16x4*)src;
                #pragma unroll
                for (int j = 0; j < 4; ++j) Bst[bcg + j][br] = v[j];
            }
            __syncthreads();

            // ---- fragments: 2x ds_load_b128 each ----
            bf16x16 afr[MI], bfr[NI];
            const int kb  = (lane >> 4) << 3;   // A: runs [kb,kb+8) and [kb+16,kb+24)
            #pragma unroll
            for (int im = 0; im < MI; ++im) {
                const bf16_t* pa = &As[wm0 + im * 16 + (lane & 15)][0];
                afr[im] = cat8(*(const bf16x8*)(pa + kb), *(const bf16x8*)(pa + kb + 16));
            }
            const int kb2 = (lane >> 4) << 3;   // B: runs [kb2,kb2+8) and [kb2+16,kb2+24)
            #pragma unroll
            for (int in = 0; in < NI; ++in) {
                const bf16_t* pb = &Bst[wn0 + in * 16 + (lane & 15)][0];
                bfr[in] = cat8(*(const bf16x8*)(pb + kb2), *(const bf16x8*)(pb + kb2 + 16));
            }
            #pragma unroll
            for (int im = 0; im < MI; ++im)
                #pragma unroll
                for (int in = 0; in < NI; ++in) {
                    if constexpr (DUAL) {
                        if (phase)
                            acc2[im][in] = __builtin_amdgcn_wmma_f32_16x16x32_bf16(
                                false, afr[im], false, bfr[in], (short)0, acc2[im][in], false, false);
                        else
                            acc[im][in] = __builtin_amdgcn_wmma_f32_16x16x32_bf16(
                                false, afr[im], false, bfr[in], (short)0, acc[im][in], false, false);
                    } else {
                        acc[im][in] = __builtin_amdgcn_wmma_f32_16x16x32_bf16(
                            false, afr[im], false, bfr[in], (short)0, acc[im][in], false, false);
                    }
                }
            __syncthreads();
        }
    }

    // ---- epilogue ----
    float*  Cfz = Cf ? Cf + z * sCb : (float*)0;
    bf16_t* Cbz = Cb ? Cb + z * sCb : (bf16_t*)0;
    const float* biasz = bias ? bias + z * sBias : (const float*)0;

    #pragma unroll
    for (int im = 0; im < MI; ++im)
        #pragma unroll
        for (int in = 0; in < NI; ++in)
            #pragma unroll
            for (int r = 0; r < 8; ++r) {
                int m = rowBase + wm0 + im * 16 + r + ((lane >> 4) << 3);
                int n = colBase + wn0 + in * 16 + (lane & 15);
                float v = acc[im][in][r];
                if constexpr (DUAL) v += alpha2 * acc2[im][in][r];
                if (biasz) v += biasz[n];
                if (pos)   v += pos[(long long)(m & 1023) * ldc + n];
                if (act == 1)      v = gelu_erf(v);
                else if (act == 2) v = softshrink(v);
                long long ci = (long long)m * ldc + n;
                if (Cfz) Cfz[ci] = v;
                if (Cbz) Cbz[ci] = (bf16_t)v;
            }
}

// =====================================================================
// Elementwise / packing kernels
// =====================================================================
__global__ void cvt_f32_bf16_kernel(const float* __restrict__ s,
                                    bf16_t* __restrict__ d, int n)
{
    int i = blockIdx.x * blockDim.x + threadIdx.x;
    if (i < n) d[i] = (bf16_t)s[i];
}

// patches (b,hp,wp) x (c,p,q) -> A0[4096][768], bf16
__global__ void pack_patches_kernel(const float* __restrict__ x,
                                    const float* __restrict__ g,
                                    bf16_t* __restrict__ A0)
{
    int i = blockIdx.x * blockDim.x + threadIdx.x;
    if (i >= NTOK * EDIM) return;
    int k = i % EDIM, m = i / EDIM;
    int b = m >> 10, hp = (m >> 5) & 31, wp = m & 31;
    int c = k >> 6, p = (k >> 3) & 7, q = k & 7;
    int h = hp * PP + p, w = wp * PP + q;
    float v = (c < TIN)
        ? x[(((long long)(b * HH + h)) * WW + w) * TIN + c]
        : g[(((long long)(b * HH + h)) * WW + w) * 2 + (c - TIN)];
    A0[i] = (bf16_t)v;
}

// conv_w (E, CIN*P*P) -> Wemb[k][e] bf16 (transpose)
__global__ void convw_tr_kernel(const float* __restrict__ s, bf16_t* __restrict__ d)
{
    int i = blockIdx.x * blockDim.x + threadIdx.x;
    if (i >= EDIM * EDIM) return;
    int e = i % EDIM, k = i / EDIM;
    d[i] = (bf16_t)s[(long long)e * EDIM + k];
}

// T (b,hp,wp,c) rows -> ls (b,f,hp,wp,c) rows, bf16
__global__ void cvt_ls_kernel(const float* __restrict__ T,
                              bf16_t* __restrict__ lsb, int fs)
{
    int i = blockIdx.x * blockDim.x + threadIdx.x;
    if (i >= NTOK * EDIM) return;
    int c = i % EDIM, m = i / EDIM;
    int b = m >> 10;
    int row = ((b * FF + fs) << 10) + (m & 1023);
    lsb[(long long)row * EDIM + c] = (bf16_t)T[i];
}

// headOut (b,f,hp,wp)[p*8+q] -> out (b, hp*8+p, wp*8+q, f)
__global__ void depatch_kernel(const float* __restrict__ ho, float* __restrict__ out)
{
    int i = blockIdx.x * blockDim.x + threadIdx.x;
    if (i >= BB * FF * HPD * WPD * PP * PP) return;
    int pq = i & 63, row = i >> 6;
    int p = pq >> 3, q = pq & 7;
    int wp = row & 31, hp = (row >> 5) & 31, fs = (row >> 10) & 3, b = row >> 12;
    out[(((long long)(b * HH + hp * PP + p)) * WW + (wp * PP + q)) * FF + fs] = ho[i];
}

// =====================================================================
// 32-point DFT kernels (ortho norm, twiddle recurrence), fp32 VALU
// =====================================================================
#define TWO_PI_32 0.19634954084936207f
#define ORTHO32   0.17677669529663687f

// rfft along wp:  T(b,hp,wp,c) -> S(b,hp,f,c) complex
__global__ void dft_w_kernel(const float* __restrict__ T,
                             float* __restrict__ Sr, float* __restrict__ Si)
{
    int i = blockIdx.x * blockDim.x + threadIdx.x;
    if (i >= NSPEC * EDIM) return;
    int c = i % EDIM;
    int t = i / EDIM;
    int f = t % WFD; t /= WFD;
    int hp = t % HPD, b = t / HPD;
    float a = -TWO_PI_32 * (float)f;
    float ca, sa; __sincosf(a, &sa, &ca);
    float cw = 1.f, sw = 0.f, sr = 0.f, si = 0.f;
    const float* row = T + (((long long)(b * HPD + hp)) * WPD) * EDIM + c;
    #pragma unroll 4
    for (int w = 0; w < WPD; ++w) {
        float v = row[(long long)w * EDIM];
        sr += v * cw; si += v * sw;
        float nc = cw * ca - sw * sa; sw = sw * ca + cw * sa; cw = nc;
    }
    Sr[i] = ORTHO32 * sr;
    Si[i] = ORTHO32 * si;
}

// complex DFT along hp: S(b,h,f,c) -> G(b,kh,f,c); sign=-1 forward, +1 inverse
__global__ void dft_h_kernel(const float* __restrict__ Sr, const float* __restrict__ Si,
                             float* __restrict__ Gr, float* __restrict__ Gi, float sign)
{
    int i = blockIdx.x * blockDim.x + threadIdx.x;
    if (i >= NSPEC * EDIM) return;
    int c = i % EDIM;
    int t = i / EDIM;
    int f = t % WFD; t /= WFD;
    int kh = t % HPD, b = t / HPD;
    float a = sign * TWO_PI_32 * (float)kh;
    float ca, sa; __sincosf(a, &sa, &ca);
    float cw = 1.f, sw = 0.f, gr = 0.f, gi = 0.f;
    const float* pr = Sr + ((long long)b * HPD * WFD + f) * EDIM + c;
    const float* pi = Si + ((long long)b * HPD * WFD + f) * EDIM + c;
    #pragma unroll 4
    for (int h = 0; h < HPD; ++h) {
        long long off = (long long)h * WFD * EDIM;
        float xr = pr[off], xi = pi[off];
        gr += xr * cw - xi * sw;
        gi += xr * sw + xi * cw;
        float nc = cw * ca - sw * sa; sw = sw * ca + cw * sa; cw = nc;
    }
    Gr[i] = ORTHO32 * gr;
    Gi[i] = ORTHO32 * gi;
}

// irfft along wp + residual add into T
__global__ void irfft_res_kernel(const float* __restrict__ Cr, const float* __restrict__ Ci,
                                 float* __restrict__ T)
{
    int i = blockIdx.x * blockDim.x + threadIdx.x;
    if (i >= NTOK * EDIM) return;
    int c = i % EDIM;
    int t = i / EDIM;
    int w = t % WPD; t /= WPD;
    int hp = t % HPD, b = t / HPD;
    float a = TWO_PI_32 * (float)w;
    float ca, sa; __sincosf(a, &sa, &ca);
    float cw = 1.f, sw = 0.f;
    const float* pr = Cr + (((long long)(b * HPD + hp)) * WFD) * EDIM + c;
    const float* pi = Ci + (((long long)(b * HPD + hp)) * WFD) * EDIM + c;
    float sum = pr[0];  // f = 0 (imag of DC dropped by C2R)
    #pragma unroll 4
    for (int f = 1; f <= 16; ++f) {
        float nc = cw * ca - sw * sa; sw = sw * ca + cw * sa; cw = nc;  // e^{i a f}
        float xr = pr[(long long)f * EDIM];
        if (f < 16) {
            float xi = pi[(long long)f * EDIM];
            sum += 2.f * (xr * cw - xi * sw);
        } else {
            sum += xr * ((w & 1) ? -1.f : 1.f);  // Nyquist: imag dropped
        }
    }
    T[i] += ORTHO32 * sum;
}

// =====================================================================
// Host orchestration
// =====================================================================
extern "C" void kernel_launch(void* const* d_in, const int* in_sizes, int n_in,
                              void* d_out, int out_size, void* d_ws, size_t ws_size,
                              hipStream_t stream)
{
    (void)in_sizes; (void)n_in; (void)out_size; (void)ws_size;

    const float* x      = (const float*)d_in[0];
    const float* grid   = (const float*)d_in[1];
    const float* conv_w = (const float*)d_in[2];
    const float* conv_b = (const float*)d_in[3];
    const float* posemb = (const float*)d_in[4];
    const float* w1     = (const float*)d_in[5];
    const float* b1     = (const float*)d_in[6];
    const float* w2     = (const float*)d_in[7];
    const float* b2     = (const float*)d_in[8];
    const float* fc1w   = (const float*)d_in[9];
    const float* fc1b_  = (const float*)d_in[10];
    const float* fc2w   = (const float*)d_in[11];
    const float* fc2b_  = (const float*)d_in[12];
    const float* hw1    = (const float*)d_in[13];
    const float* hb1    = (const float*)d_in[14];
    const float* hw2    = (const float*)d_in[15];
    const float* hb2    = (const float*)d_in[16];
    float* out = (float*)d_out;

    // ---- bump allocator over workspace ----
    char* wp_ = (char*)d_ws;
    auto alloc = [&](size_t bytes) -> void* {
        void* r = (void*)wp_;
        wp_ += (bytes + 255) & ~(size_t)255;
        return r;
    };
    const int nT = NTOK * EDIM;    // 3,145,728
    const int nS = NSPEC * EDIM;   // 1,671,168

    float*  T    = (float*)alloc((size_t)nT * 4);
    float*  Ar   = (float*)alloc((size_t)nS * 4);
    float*  Ai   = (float*)alloc((size_t)nS * 4);
    float*  Br   = (float*)alloc((size_t)nS * 4);
    float*  Bi   = (float*)alloc((size_t)nS * 4);
    bf16_t* gbr  = (bf16_t*)alloc((size_t)nS * 2);
    bf16_t* gbi  = (bf16_t*)alloc((size_t)nS * 2);
    bf16_t* h1br = (bf16_t*)alloc((size_t)nS * 2);
    bf16_t* h1bi = (bf16_t*)alloc((size_t)nS * 2);
    bf16_t* tb   = (bf16_t*)alloc((size_t)nT * 2);
    bf16_t* hidb = (bf16_t*)alloc((size_t)NTOK * MIDD * 2);
    bf16_t* lsb  = (bf16_t*)alloc((size_t)NTOK * FF * EDIM * 2);
    bf16_t* hhb  = (bf16_t*)alloc((size_t)NTOK * FF * (2 * EDIM) * 2);
    float*  hout = (float*)alloc((size_t)NTOK * FF * (PP * PP) * 4);
    bf16_t* A0   = (bf16_t*)alloc((size_t)nT * 2);
    bf16_t* wembb= (bf16_t*)alloc((size_t)EDIM * EDIM * 2);
    bf16_t* w1b  = (bf16_t*)alloc((size_t)DD * 2 * NBLK * BSZ * BSZ * 2);
    bf16_t* w2b  = (bf16_t*)alloc((size_t)DD * 2 * NBLK * BSZ * BSZ * 2);
    bf16_t* fc1bw= (bf16_t*)alloc((size_t)DD * EDIM * MIDD * 2);
    bf16_t* fc2bw= (bf16_t*)alloc((size_t)DD * MIDD * EDIM * 2);
    bf16_t* hw1b = (bf16_t*)alloc((size_t)EDIM * 2 * EDIM * 2);
    bf16_t* hw2b = (bf16_t*)alloc((size_t)2 * EDIM * PP * PP * 2);

    auto cvt = [&](const float* s, bf16_t* d, int n) {
        cvt_f32_bf16_kernel<<<dim3((n + 255) / 256), dim3(256), 0, stream>>>(s, d, n);
    };
    // big single GEMM: block tile 128x64, wave tile 32x32
    auto gemm_big = [&](const bf16_t* A, int lda, const bf16_t* Bm, int ldb,
                        float* Cf, bf16_t* Cb, int ldc,
                        const float* bias, const float* pos,
                        int M, int N, int K, int act) {
        dim3 g(N / 64, M / 128, 1);
        wmma_gemm_kernel<32, 32, false><<<g, dim3(256), 0, stream>>>(
            A, (const bf16_t*)0, 0, lda, Bm, (const bf16_t*)0, 0, ldb, 0.f,
            Cf, Cb, 0, ldc, bias, 0, pos, K, act);
    };
    // fused dual GEMM over the 8 diagonal blocks (complex MLP):
    //   C = A1@B1 + alpha2 * A2@B2 + bias -> act
    auto gemm_blk = [&](const bf16_t* A1, const bf16_t* A2,
                        const bf16_t* B1, const bf16_t* B2, float alpha2,
                        float* Cf, bf16_t* Cb, const float* bias, int act) {
        dim3 g(BSZ / 32, NSPEC / 64, NBLK);
        wmma_gemm_kernel<16, 16, true><<<g, dim3(256), 0, stream>>>(
            A1, A2, (long long)BSZ, EDIM, B1, B2, (long long)BSZ * BSZ, BSZ, alpha2,
            Cf, Cb, (long long)BSZ, EDIM, bias, (long long)BSZ, (const float*)0, BSZ, act);
    };

    // ---- weight conversions (bf16) ----
    const int nW1 = DD * 2 * NBLK * BSZ * BSZ;       // 589,824
    cvt(w1, w1b, nW1);
    cvt(w2, w2b, nW1);
    cvt(fc1w, fc1bw, DD * EDIM * MIDD);
    cvt(fc2w, fc2bw, DD * MIDD * EDIM);
    cvt(hw1, hw1b, EDIM * 2 * EDIM);
    cvt(hw2, hw2b, 2 * EDIM * PP * PP);
    convw_tr_kernel<<<dim3((EDIM * EDIM + 255) / 256), dim3(256), 0, stream>>>(conv_w, wembb);
    pack_patches_kernel<<<dim3((nT + 255) / 256), dim3(256), 0, stream>>>(x, grid, A0);

    // ---- patch embedding: T = A0 @ Wemb + conv_b + pos_emb ----
    gemm_big(A0, EDIM, wembb, EDIM, T, 0, EDIM, conv_b, posemb, NTOK, EDIM, EDIM, 0);

    const dim3 gS((nS + 255) / 256), gT((nT + 255) / 256), b256(256);
    const long long wblk = (long long)NBLK * BSZ * BSZ;  // 73728 per (d, re/im)

    for (int fs = 0; fs < FF; ++fs) {
        for (int d = 0; d < DD; ++d) {
            // ---------- spectral block ----------
            dft_w_kernel<<<gS, b256, 0, stream>>>(T, Ar, Ai);
            dft_h_kernel<<<gS, b256, 0, stream>>>(Ar, Ai, Br, Bi, -1.f);
            cvt(Br, gbr, nS);
            cvt(Bi, gbi, nS);

            const bf16_t* W1r = w1b + (long long)(d * 2 + 0) * wblk;
            const bf16_t* W1i = w1b + (long long)(d * 2 + 1) * wblk;
            const bf16_t* W2r = w2b + (long long)(d * 2 + 0) * wblk;
            const bf16_t* W2i = w2b + (long long)(d * 2 + 1) * wblk;
            const float* b1r = b1 + (long long)(d * 2 + 0) * NBLK * BSZ;
            const float* b1i = b1 + (long long)(d * 2 + 1) * NBLK * BSZ;
            const float* b2r = b2 + (long long)(d * 2 + 0) * NBLK * BSZ;
            const float* b2i = b2 + (long long)(d * 2 + 1) * NBLK * BSZ;

            // o1r = gelu(xr@W1r - xi@W1i + b1r)   -> bf16 directly
            gemm_blk(gbr, gbi, W1r, W1i, -1.f, (float*)0, h1br, b1r, 1);
            // o1i = gelu(xi@W1r + xr@W1i + b1i)
            gemm_blk(gbi, gbr, W1r, W1i, +1.f, (float*)0, h1bi, b1i, 1);
            // o2r = shrink(o1r@W2r - o1i@W2i + b2r) -> f32 for inverse DFT
            gemm_blk(h1br, h1bi, W2r, W2i, -1.f, Br, (bf16_t*)0, b2r, 2);
            // o2i = shrink(o1i@W2r + o1r@W2i + b2i)
            gemm_blk(h1bi, h1br, W2r, W2i, +1.f, Bi, (bf16_t*)0, b2i, 2);

            dft_h_kernel<<<gS, b256, 0, stream>>>(Br, Bi, Ar, Ai, +1.f);   // ifft along h
            irfft_res_kernel<<<gT, b256, 0, stream>>>(Ar, Ai, T);          // irfft + residual

            // ---------- channel MLP ----------
            cvt(T, tb, nT);
            gemm_big(tb, EDIM, fc1bw + (long long)d * EDIM * MIDD, MIDD,
                     (float*)0, hidb, MIDD, fc1b_ + (long long)d * MIDD, (const float*)0,
                     NTOK, MIDD, EDIM, 1);
            gemm_big(hidb, MIDD, fc2bw + (long long)d * MIDD * EDIM, EDIM,
                     T, (bf16_t*)0, EDIM, fc2b_ + (long long)d * EDIM, (const float*)0,
                     NTOK, EDIM, MIDD, 0);
        }
        cvt_ls_kernel<<<gT, b256, 0, stream>>>(T, lsb, fs);
    }

    // ---- head: gelu(ls @ Hw1 + hb1) @ Hw2 + hb2 ----
    gemm_big(lsb, EDIM, hw1b, 2 * EDIM, (float*)0, hhb, 2 * EDIM,
             hb1, (const float*)0, NTOK * FF, 2 * EDIM, EDIM, 1);
    gemm_big(hhb, 2 * EDIM, hw2b, PP * PP, hout, (bf16_t*)0, PP * PP,
             hb2, (const float*)0, NTOK * FF, PP * PP, 2 * EDIM, 0);

    depatch_kernel<<<dim3((NTOK * FF * PP * PP + 255) / 256), dim3(256), 0, stream>>>(hout, out);
}
